// MultiHeadAttention_76321568850683
// MI455X (gfx1250) — matrix operand order repaired and checked by reading, not compiled, not run on hardware
//
#include <hip/hip_runtime.h>

typedef _Float16 f16;
typedef __attribute__((ext_vector_type(16))) _Float16 v16h;
typedef __attribute__((ext_vector_type(8)))  float    v8f;

union AF { v16h v; f16 h[16]; uint4 u4[2]; };
union CF { v8f v; float f[8]; };

__device__ __forceinline__ v8f wmma_f16(v16h a, v16h b, v8f c) {
  // (neg_a, A, neg_b, B, c_mod, C, reuse_a, reuse_b)
  return __builtin_amdgcn_wmma_f32_16x16x32_f16(false, a, false, b, (short)0, c, false, false);
}

// CDNA5 async global->LDS path (ASYNCcnt-tracked), guarded so a toolchain
// without the builtins falls back to synchronous uint4 copies.
#if defined(__gfx1250__) && __has_builtin(__builtin_amdgcn_global_load_async_to_lds_b128) && __has_builtin(__builtin_amdgcn_s_wait_asynccnt)
#define USE_ASYNC_LDS 1
typedef int v4i_b128 __attribute__((vector_size(16)));
typedef __attribute__((address_space(1))) v4i_b128 gv4i;   // global int4
typedef __attribute__((address_space(3))) v4i_b128 lv4i;   // LDS int4
#else
#define USE_ASYNC_LDS 0
#endif

// ---------------------------------------------------------------------------
// Kernel 1: QKV projection. C(M x 128) = x(M x 128) @ W(128 x 128), f16 out,
// written head-major: dst[(bt*H + h)*288*32 + tok*32 + d].
// grid = (M/16, 128/64, 3), block = 128 threads (4 waves, each a 16x16 tile).
// ---------------------------------------------------------------------------
__global__ __launch_bounds__(128) void qkv_gemm_kernel(
    const float* __restrict__ x,
    const float* __restrict__ Wq, const float* __restrict__ Wk,
    const float* __restrict__ Wv,
    f16* __restrict__ q16, f16* __restrict__ k16, f16* __restrict__ v16)
{
  __shared__ __align__(16) f16 As[16 * 128];   // [m][k]
  __shared__ __align__(16) f16 Wt[64 * 128];   // [n_local][k] (transposed)

  const int tid  = threadIdx.x;
  const int lane = tid & 31, wid = tid >> 5;
  const int row0 = blockIdx.x * 16;
  const int col0 = blockIdx.y * 64;
  const float* W = (blockIdx.z == 0) ? Wq : (blockIdx.z == 1) ? Wk : Wv;
  f16*       dst = (blockIdx.z == 0) ? q16 : (blockIdx.z == 1) ? k16 : v16;

  // Stage A tile: 16 rows x 128 ch, f32 -> f16 (512 float4)
  for (int i = tid; i < 512; i += 128) {
    int r = i >> 5, c4 = i & 31;
    float4 f = ((const float4*)(x + (size_t)(row0 + r) * 128))[c4];
    f16* p = &As[r * 128 + c4 * 4];
    p[0] = (f16)f.x; p[1] = (f16)f.y; p[2] = (f16)f.z; p[3] = (f16)f.w;
  }
  // Stage W tile transposed: rows k=0..127, cols col0..col0+63 (2048 float4)
  for (int i = tid; i < 2048; i += 128) {
    int k = i >> 4, c4 = i & 15;
    float4 f = ((const float4*)(W + (size_t)k * 128 + col0))[c4];
    Wt[(c4 * 4 + 0) * 128 + k] = (f16)f.x;
    Wt[(c4 * 4 + 1) * 128 + k] = (f16)f.y;
    Wt[(c4 * 4 + 2) * 128 + k] = (f16)f.z;
    Wt[(c4 * 4 + 3) * 128 + k] = (f16)f.w;
  }
  __syncthreads();

  const int hf = lane >> 4, nl = lane & 15;
  const int ncl = wid * 16 + nl;           // local column 0..63

  // Load all fragments first (distinct regs), then back-to-back WMMAs.
  AF a[4], b[4];
  #pragma unroll
  for (int kk = 0; kk < 4; ++kk) {
    const int k0 = kk * 32;
    a[kk].u4[0] = *(const uint4*)&As[nl * 128 + k0 + hf * 8];
    a[kk].u4[1] = *(const uint4*)&As[nl * 128 + k0 + 16 + hf * 8];
    b[kk].u4[0] = *(const uint4*)&Wt[ncl * 128 + k0 + hf * 8];
    b[kk].u4[1] = *(const uint4*)&Wt[ncl * 128 + k0 + 16 + hf * 8];
  }
  CF c;
  #pragma unroll
  for (int i = 0; i < 8; ++i) c.f[i] = 0.f;
  #pragma unroll
  for (int kk = 0; kk < 4; ++kk) c.v = wmma_f16(a[kk].v, b[kk].v, c.v);

  // Epilogue: split heads, write f16. All 16 rows share bt (288 % 16 == 0),
  // so hoist the div/mod to one uniform computation.
  const int col = col0 + ncl;
  const int h = col >> 5, d = col & 31;
  const int bt = row0 / 288;               // uniform -> scalar
  const int tok0 = row0 % 288;             // uniform -> scalar
  f16* dptr = dst + (((size_t)bt * 4 + h) * 288 + tok0 + hf * 8) * 32 + d;
  #pragma unroll
  for (int v = 0; v < 8; ++v) dptr[(size_t)v * 32] = (f16)c.f[v];
}

// ---------------------------------------------------------------------------
// Kernel 2: fused attention per (bt, head). grid = 512 blocks x 64 threads.
// K, V^T resident in LDS; each wave handles 16-query tiles.
// ---------------------------------------------------------------------------
__global__ __launch_bounds__(64) void attn_kernel(
    const f16* __restrict__ q16, const f16* __restrict__ k16,
    const f16* __restrict__ v16, const float* __restrict__ drop_u,
    const int* __restrict__ nxp, float* __restrict__ ctx)
{
  __shared__ __align__(16) f16 Ks[288 * 32];      // [key][ch]
  __shared__ __align__(16) f16 Vt[32 * 288];      // [ch][key] (transposed)
  __shared__ __align__(16) f16 Sh[2][16 * 288];   // per-wave weight tile

  const int tid  = threadIdx.x;
  const int lane = tid & 31, wid = tid >> 5;
  const int bh   = blockIdx.x;                    // bt*4 + h
  const size_t base = (size_t)bh * 288 * 32;

  // Stage K: async global->LDS b128 (overlaps the manual V transpose below).
#if USE_ASYNC_LDS
  for (int i = tid; i < 1152; i += 64) {
    __builtin_amdgcn_global_load_async_to_lds_b128(
        (gv4i*)(k16 + base) + i, (lv4i*)Ks + i, 0, 0);
  }
#else
  {
    const uint4* ks = (const uint4*)(k16 + base);
    uint4* kd = (uint4*)Ks;
    for (int i = tid; i < 1152; i += 64) kd[i] = ks[i];
  }
#endif
  // Stage V transposed (manual: needs reshuffle during staging).
  {
    const uint4* vs = (const uint4*)(v16 + base);
    for (int i = tid; i < 1152; i += 64) {
      uint4 u = vs[i];
      int key = i >> 2, cb = (i & 3) * 8;
      const f16* hv = (const f16*)&u;
      #pragma unroll
      for (int j = 0; j < 8; ++j) Vt[(cb + j) * 288 + key] = hv[j];
    }
  }
#if USE_ASYNC_LDS
  __builtin_amdgcn_s_wait_asynccnt(0);
#endif
  __syncthreads();

  const int nx = *nxp;
  const float keep_x = 0.9f, keep_lq = 0.7f;
  const float nxx  = (float)nx * (float)nx;
  const float ntot = 288.f * 288.f;
  const float scale = ntot / (nxx * keep_x + (ntot - nxx) * keep_lq);
  const float rsH = 0.17677669529663687f;        // 1/sqrt(32)

  const int hf = lane >> 4, nl = lane & 15;
  f16* myS = Sh[wid];
  const int bt = bh >> 2, h = bh & 3;

  auto load_bk = [&](AF& f, int t) {
    const f16* kp = &Ks[(t * 16 + nl) * 32 + hf * 8];
    f.u4[0] = *(const uint4*)(kp);
    f.u4[1] = *(const uint4*)(kp + 16);
  };
  auto load_aw = [&](AF& f, int kb) {
    const f16* wp = &myS[nl * 288 + kb * 32];
    f.u4[0] = *(const uint4*)(wp + hf * 8);
    f.u4[1] = *(const uint4*)(wp + 16 + hf * 8);
  };
  auto load_bv = [&](AF& f, int kb, int nt) {
    const f16* vp = &Vt[(nt * 16 + nl) * 288 + kb * 32 + hf * 8];
    f.u4[0] = *(const uint4*)(vp);
    f.u4[1] = *(const uint4*)(vp + 16);
  };

  for (int qt = wid; qt < 18; qt += 2) {         // wave-uniform loop
    // Q A-fragment (16 rows x K=32) straight from global f16
    AF aq;
    const f16* qrowp = q16 + base + (size_t)(qt * 16 + nl) * 32;
    aq.u4[0] = *(const uint4*)(qrowp + hf * 8);
    aq.u4[1] = *(const uint4*)(qrowp + 16 + hf * 8);

    // Scores: S = Q x K^T over 18 key tiles (double-buffered B fragments)
    {
      AF bk[2];
      load_bk(bk[0], 0);
      for (int t = 0; t < 18; ++t) {
        if (t + 1 < 18) load_bk(bk[(t + 1) & 1], t + 1);
        CF c;
        #pragma unroll
        for (int i = 0; i < 8; ++i) c.f[i] = 0.f;
        c.v = wmma_f16(aq.v, bk[t & 1].v, c.v);
        #pragma unroll
        for (int v = 0; v < 8; ++v)
          myS[(hf * 8 + v) * 288 + t * 16 + nl] = (f16)(c.f[v] * rsH);
      }
    }

    // Softmax phase A: row r = nl; lane-half hf covers cols [hf*144, +144).
    // Compute row max, exp (cached in place), and row sum.
    float inv;
    {
      const int r = nl;
      f16* srow = &myS[r * 288];
      const int c0 = hf * 144, c1 = c0 + 144;
      float mx = -3.0e38f;
      for (int cc = c0; cc < c1; ++cc) mx = fmaxf(mx, (float)srow[cc]);
      mx = fmaxf(mx, __shfl_xor(mx, 16, 32));
      float sm = 0.f;
      for (int cc = c0; cc < c1; ++cc) {
        float e = __expf((float)srow[cc] - mx);
        sm += e;
        srow[cc] = (f16)e;                        // cache exp in place
      }
      sm += __shfl_xor(sm, 16, 32);
      inv = 1.0f / sm;                            // lane r holds row r's inv
    }

    // Softmax phase B: normalize + dropout, fully coalesced drop_u stream.
    // All 32 lanes sweep one row together: c = j*32 + lane -> each load
    // instruction covers 32 consecutive floats (one 128B transaction).
    for (int r = 0; r < 16; ++r) {
      const float invr = __shfl(inv, r, 32);      // broadcast row r's 1/sum
      const int qrow = qt * 16 + r;
      f16* srow = &myS[r * 288];
      const float* du = drop_u + ((size_t)bh * 288 + qrow) * 288;
      const bool qx = qrow < nx;
      #pragma unroll 3
      for (int j = 0; j < 9; ++j) {
        const int cc = j * 32 + lane;
        __builtin_prefetch(du + cc + 288, 0, 0);  // next row ahead
        float w  = (float)srow[cc] * invr;
        float kp = (qx && cc < nx) ? keep_x : keep_lq;
        float m  = (du[cc] < kp) ? scale : 0.f;
        srow[cc] = (f16)(w * m);
      }
    }

    // ctx = W(16x288) x V(288x32): 2 column tiles x 9 K-chunks of 32
    #pragma unroll
    for (int nt = 0; nt < 2; ++nt) {
      CF c;
      #pragma unroll
      for (int i = 0; i < 8; ++i) c.f[i] = 0.f;
      AF aw[2], bv[2];
      load_aw(aw[0], 0);
      load_bv(bv[0], 0, nt);
      for (int kb = 0; kb < 9; ++kb) {
        if (kb + 1 < 9) {
          load_aw(aw[(kb + 1) & 1], kb + 1);
          load_bv(bv[(kb + 1) & 1], kb + 1, nt);
        }
        c.v = wmma_f16(aw[kb & 1].v, bv[kb & 1].v, c.v);
      }
      #pragma unroll
      for (int v = 0; v < 8; ++v) {
        int qrow = qt * 16 + hf * 8 + v;
        ctx[((size_t)bt * 288 + qrow) * 128 + h * 32 + nt * 16 + nl] = c.f[v];
      }
    }
  }
}

// ---------------------------------------------------------------------------
// Kernel 3: out = ctx(M x 128) @ Wo(128 x 128) + bo, f32 out.
// grid = (M/16, 2), block = 128 threads.
// ---------------------------------------------------------------------------
__global__ __launch_bounds__(128) void out_gemm_kernel(
    const float* __restrict__ A, const float* __restrict__ Wo,
    const float* __restrict__ bo, float* __restrict__ out)
{
  __shared__ __align__(16) f16 As[16 * 128];
  __shared__ __align__(16) f16 Wt[64 * 128];

  const int tid  = threadIdx.x;
  const int lane = tid & 31, wid = tid >> 5;
  const int row0 = blockIdx.x * 16;
  const int col0 = blockIdx.y * 64;

  for (int i = tid; i < 512; i += 128) {
    int r = i >> 5, c4 = i & 31;
    float4 f = ((const float4*)(A + (size_t)(row0 + r) * 128))[c4];
    f16* p = &As[r * 128 + c4 * 4];
    p[0] = (f16)f.x; p[1] = (f16)f.y; p[2] = (f16)f.z; p[3] = (f16)f.w;
  }
  for (int i = tid; i < 2048; i += 128) {
    int k = i >> 4, c4 = i & 15;
    float4 f = ((const float4*)(Wo + (size_t)k * 128 + col0))[c4];
    Wt[(c4 * 4 + 0) * 128 + k] = (f16)f.x;
    Wt[(c4 * 4 + 1) * 128 + k] = (f16)f.y;
    Wt[(c4 * 4 + 2) * 128 + k] = (f16)f.z;
    Wt[(c4 * 4 + 3) * 128 + k] = (f16)f.w;
  }
  __syncthreads();

  const int hf = lane >> 4, nl = lane & 15;
  const int ncl = wid * 16 + nl;

  AF a[4], b[4];
  #pragma unroll
  for (int kk = 0; kk < 4; ++kk) {
    const int k0 = kk * 32;
    a[kk].u4[0] = *(const uint4*)&As[nl * 128 + k0 + hf * 8];
    a[kk].u4[1] = *(const uint4*)&As[nl * 128 + k0 + 16 + hf * 8];
    b[kk].u4[0] = *(const uint4*)&Wt[ncl * 128 + k0 + hf * 8];
    b[kk].u4[1] = *(const uint4*)&Wt[ncl * 128 + k0 + 16 + hf * 8];
  }
  CF c;
  #pragma unroll
  for (int i = 0; i < 8; ++i) c.f[i] = 0.f;
  #pragma unroll
  for (int kk = 0; kk < 4; ++kk) c.v = wmma_f16(a[kk].v, b[kk].v, c.v);

  const int col = col0 + ncl;
  const float bias = bo[col];
  float* optr = out + (size_t)(row0 + hf * 8) * 128 + col;
  #pragma unroll
  for (int v = 0; v < 8; ++v) optr[(size_t)v * 128] = c.f[v] + bias;
}

// ---------------------------------------------------------------------------
extern "C" void kernel_launch(void* const* d_in, const int* in_sizes, int n_in,
                              void* d_out, int out_size, void* d_ws, size_t ws_size,
                              hipStream_t stream) {
  const float* x  = (const float*)d_in[0];
  const float* Wq = (const float*)d_in[1];
  const float* Wk = (const float*)d_in[2];
  const float* Wv = (const float*)d_in[3];
  const float* Wo = (const float*)d_in[4];
  const float* bo = (const float*)d_in[5];
  const float* du = (const float*)d_in[6];
  const int*   nx = (const int*)d_in[7];
  float* out = (float*)d_out;

  const size_t M = 4ull * 32ull * 288ull;   // 36864 tokens
  f16*  q16 = (f16*)d_ws;
  f16*  k16 = q16 + M * 128;
  f16*  v16 = k16 + M * 128;
  float* ctx = (float*)(v16 + M * 128);

  qkv_gemm_kernel<<<dim3((unsigned)(M / 16), 2, 3), dim3(128), 0, stream>>>(
      x, Wq, Wk, Wv, q16, k16, v16);
  attn_kernel<<<dim3(512), dim3(64), 0, stream>>>(q16, k16, v16, du, nx, ctx);
  out_gemm_kernel<<<dim3((unsigned)(M / 16), 2, 1), dim3(128), 0, stream>>>(
      ctx, Wo, bo, out);
}
